// HeteroSTBlock_25005299597589
// MI455X (gfx1250) — matrix operand order repaired and checked by reading, not compiled, not run on hardware
//
#include <hip/hip_runtime.h>
#include <cmath>

// ---------------------------------------------------------------------------
// CDNA5 (gfx1250) wave32 WMMA, v_wmma_f32_16x16x32_f16.
// Fragments are staged in LDS **in fragment order**: for each 32-wide K step,
// lane L's 16 f16 elements are contiguous (32B), so a fragment load is one
// 128-bit x2 LDS read instead of 16 scalar ds_load_u16.
// Mapping (ISA 7.12.2):
//   A: lane -> m = L&15, half = L>>4; elem e -> v=e>>1, odd=e&1;
//      k = (v<4 ? 2v : 16+2(v-4)) + 8*half + odd
//   B: lane -> n = L&15, half = L>>4; k = same as A
//   C/D: elem r -> m = r + 8*(lane>>4), n = lane&15
// ---------------------------------------------------------------------------
typedef __attribute__((ext_vector_type(16))) _Float16 v16h;
typedef __attribute__((ext_vector_type(8)))  float    v8f;

#define WMMA_F16(a, b, c) \
  __builtin_amdgcn_wmma_f32_16x16x32_f16(false, (a), false, (b), (short)0, (c), false, false)

__device__ __forceinline__ float dsigmoid(float x) { return 1.0f / (1.0f + expf(-x)); }

// destination-driven index math for fragment staging
__device__ __forceinline__ int a_k_of(int lane_d, int e) {
  const int half = lane_d >> 4, v = e >> 1, odd = e & 1;
  return ((v < 4) ? (2 * v) : (16 + 2 * (v - 4))) + 8 * half + odd;
}
__device__ __forceinline__ int b_k_of(int lane_d, int e) {
  const int half = lane_d >> 4, v = e >> 1, odd = e & 1;
  return ((v < 4) ? (2 * v) : (16 + 2 * (v - 4))) + 8 * half + odd;
}

__device__ __forceinline__ void store_D_lds(float* D, int ldn, v8f d, int lane) {
  const int n = lane & 15, half = lane >> 4;
#pragma unroll
  for (int r = 0; r < 8; ++r) D[(size_t)(r + 8 * half) * ldn + n] = d[r];
}

// ---------------------------------------------------------------------------
// tconv #1: X (B,Cin,T,N) -> flat [t*B*N + b*N + n, CH], gated residual.
// 16 rows/block, 128 threads (4 waves). All shape params compile-time.
// ---------------------------------------------------------------------------
template <int CIN, int CH, int KPAD>
__global__ __launch_bounds__(128) void tconv1_kernel(
    const float* __restrict__ X, const float* __restrict__ W,
    const float* __restrict__ bias, float* __restrict__ Y,
    int Bn, int T, int N, int P) {
  constexpr int K = CIN * 3;
  constexpr int TWOCH = 2 * CH;
  constexpr int NTILES = TWOCH / 16;
  constexpr int NTPW = (NTILES + 3) / 4;
  constexpr int KSTEPS = KPAD / 32;
  __shared__ __align__(32) _Float16 AsF[KSTEPS * 512];
  __shared__ __align__(32) _Float16 BsF[2 * NTILES * 512];
  __shared__ float Yt[16 * TWOCH];
  const int tid = threadIdx.x, lane = tid & 31, wave = tid >> 5;
  const int row0 = blockIdx.x * 16;
  const int BN = Bn * N;

  // ---- stage all A fragments (fragment order) ----
  for (int i = tid; i < KSTEPS * 512; i += 128) {
    const int s = i >> 9, d = i & 511, lane_d = d >> 4, e = d & 15;
    const int m = lane_d & 15;
    const int kk = s * 32 + a_k_of(lane_d, e);
    float v = 0.f;
    const int pos = row0 + m;
    if (pos < P && kk < K) {
      const int c = kk / 3, kt = kk - 3 * c;
      const int t = pos / BN, rem = pos - t * BN, b = rem / N, n = rem - b * N;
      const float* src = &X[((b * CIN + c) * T + (t + kt)) * N + n];
      __builtin_prefetch(src + N, 0, 1);  // global_prefetch_b8
      v = *src;
    }
    AsF[i] = (_Float16)v;
  }

  v8f acc[NTPW] = {};
#pragma unroll
  for (int kc = 0; kc < KPAD; kc += 64) {
    // ---- stage B fragments for this 64-wide K chunk ----
    for (int i = tid; i < 2 * NTILES * 512; i += 128) {
      const int s = i / (NTILES * 512);
      const int rem = i - s * (NTILES * 512);
      const int nt = rem >> 9, d = rem & 511, lane_d = d >> 4, e = d & 15;
      const int kk = kc + s * 32 + b_k_of(lane_d, e);
      const int o = nt * 16 + (lane_d & 15);
      float v = 0.f;
      if (kk < K) { const int c = kk / 3, kt = kk - 3 * c; v = W[(o * CIN + c) * 3 + kt]; }
      BsF[i] = (_Float16)v;
    }
    __syncthreads();
#pragma unroll
    for (int s = 0; s < 2; ++s) {
      if (kc + s * 32 < KPAD) {
        const v16h a = *reinterpret_cast<const v16h*>(&AsF[((kc >> 5) + s) * 512 + lane * 16]);
#pragma unroll
        for (int ai = 0; ai < NTPW; ++ai) {
          const int nt = wave + ai * 4;
          if (nt < NTILES) {
            const v16h bf = *reinterpret_cast<const v16h*>(&BsF[(s * NTILES + nt) * 512 + lane * 16]);
            acc[ai] = WMMA_F16(a, bf, acc[ai]);
          }
        }
      }
    }
    __syncthreads();
  }
#pragma unroll
  for (int ai = 0; ai < NTPW; ++ai) {
    const int nt = wave + ai * 4;
    if (nt < NTILES) store_D_lds(Yt + nt * 16, TWOCH, acc[ai], lane);
  }
  __syncthreads();
  // ---- gate + residual + store flat ----
  for (int i = tid; i < 16 * CH; i += 128) {
    const int r = i / CH, o = i - r * CH;
    const int pos = row0 + r;
    if (pos >= P) continue;
    const int t = pos / BN, rem = pos - t * BN, b = rem / N, n = rem - b * N;
    const float pv = Yt[r * TWOCH + o] + bias[o];
    const float qv = Yt[r * TWOCH + CH + o] + bias[CH + o];
    const float xin = (o < CIN) ? X[((b * CIN + o) * T + (t + 2)) * N + n] : 0.f;
    Y[(size_t)pos * CH + o] = (pv + xin) * dsigmoid(qv);
  }
}

// ---------------------------------------------------------------------------
// tconv #2 + LayerNorm: X flat [P1, CM] (t stride B*N) -> out (B,64,T2,N).
// ---------------------------------------------------------------------------
template <int CM, int KPAD>
__global__ __launch_bounds__(128) void tconv2_ln_kernel(
    const float* __restrict__ Xf, const float* __restrict__ W,
    const float* __restrict__ bias, const float* __restrict__ lng,
    const float* __restrict__ lnb, float* __restrict__ out,
    int Bn, int N, int T2, int P) {
  constexpr int K = CM * 3;
  constexpr int NTILES = 8;  // 2*64 output channels
  constexpr int KSTEPS = KPAD / 32;
  __shared__ __align__(32) _Float16 AsF[KSTEPS * 512];
  __shared__ __align__(32) _Float16 BsF[2 * NTILES * 512];
  __shared__ float Yt[16 * 128];
  __shared__ float Ot[16 * 64];
  __shared__ float mu_s[16], rs_s[16];
  const int tid = threadIdx.x, lane = tid & 31, wave = tid >> 5;
  const int row0 = blockIdx.x * 16;
  const int BN = Bn * N;

  for (int i = tid; i < KSTEPS * 512; i += 128) {
    const int s = i >> 9, d = i & 511, lane_d = d >> 4, e = d & 15;
    const int m = lane_d & 15;
    const int kk = s * 32 + a_k_of(lane_d, e);
    float v = 0.f;
    const int pos = row0 + m;
    if (pos < P && kk < K) {
      const int c = kk / 3, kt = kk - 3 * c;
      v = Xf[(size_t)(pos + kt * BN) * CM + c];
    }
    AsF[i] = (_Float16)v;
  }

  v8f acc[2] = {};
#pragma unroll
  for (int kc = 0; kc < KPAD; kc += 64) {
    for (int i = tid; i < 2 * NTILES * 512; i += 128) {
      const int s = i >> 12;           // NTILES*512 == 4096
      const int rem = i & 4095;
      const int nt = rem >> 9, d = rem & 511, lane_d = d >> 4, e = d & 15;
      const int kk = kc + s * 32 + b_k_of(lane_d, e);
      const int o = nt * 16 + (lane_d & 15);
      float v = 0.f;
      if (kk < K) { const int c = kk / 3, kt = kk - 3 * c; v = W[(o * CM + c) * 3 + kt]; }
      BsF[i] = (_Float16)v;
    }
    __syncthreads();
#pragma unroll
    for (int s = 0; s < 2; ++s) {
      if (kc + s * 32 < KPAD) {
        const v16h a = *reinterpret_cast<const v16h*>(&AsF[((kc >> 5) + s) * 512 + lane * 16]);
#pragma unroll
        for (int ai = 0; ai < 2; ++ai) {
          const int nt = wave + ai * 4;
          const v16h bf = *reinterpret_cast<const v16h*>(&BsF[(s * NTILES + nt) * 512 + lane * 16]);
          acc[ai] = WMMA_F16(a, bf, acc[ai]);
        }
      }
    }
    __syncthreads();
  }
#pragma unroll
  for (int ai = 0; ai < 2; ++ai) store_D_lds(Yt + (wave + ai * 4) * 16, 128, acc[ai], lane);
  __syncthreads();
  for (int i = tid; i < 16 * 64; i += 128) {
    const int r = i >> 6, o = i & 63;
    const float pv = Yt[r * 128 + o] + bias[o];
    const float qv = Yt[r * 128 + 64 + o] + bias[64 + o];
    const int pos = row0 + r;
    float xin = 0.f;
    if (pos < P && o < CM) xin = Xf[(size_t)(pos + 2 * BN) * CM + o];
    Ot[i] = (pv + xin) * dsigmoid(qv);
  }
  __syncthreads();
  if (tid < 16) {
    float m = 0.f;
    for (int o = 0; o < 64; ++o) m += Ot[tid * 64 + o];
    m *= (1.f / 64.f);
    float var = 0.f;
    for (int o = 0; o < 64; ++o) { const float dd = Ot[tid * 64 + o] - m; var += dd * dd; }
    var *= (1.f / 64.f);
    mu_s[tid] = m;
    rs_s[tid] = rsqrtf(var + 1e-5f);
  }
  __syncthreads();
  for (int i = tid; i < 16 * 64; i += 128) {
    const int r = i >> 6, o = i & 63;
    const int pos = row0 + r;
    if (pos >= P) continue;
    const int t = pos / BN, rem = pos - t * BN, b = rem / N, n = rem - b * N;
    out[((b * 64 + o) * T2 + t) * N + n] = (Ot[i] - mu_s[r]) * rs_s[r] * lng[o] + lnb[o];
  }
}

// ---------------------------------------------------------------------------
// Generic Y[rows,64] = X[rows,K] @ Wt[64,K]^T (+bias). K <= 64.
// ---------------------------------------------------------------------------
__global__ __launch_bounds__(128) void gemm_nt_bias_kernel(
    const float* __restrict__ X, const float* __restrict__ Wt,
    const float* __restrict__ bias, float* __restrict__ Y, int rows, int K) {
  __shared__ __align__(32) _Float16 AsF[2 * 512];
  __shared__ __align__(32) _Float16 BsF[2 * 4 * 512];
  const int tid = threadIdx.x, lane = tid & 31, wave = tid >> 5;
  const int row0 = blockIdx.x * 16;
  for (int i = tid; i < 2 * 512; i += 128) {
    const int s = i >> 9, d = i & 511, lane_d = d >> 4, e = d & 15;
    const int m = lane_d & 15;
    const int kk = s * 32 + a_k_of(lane_d, e);
    float v = 0.f;
    if (row0 + m < rows && kk < K) v = X[(size_t)(row0 + m) * K + kk];
    AsF[i] = (_Float16)v;
  }
  for (int i = tid; i < 2 * 4 * 512; i += 128) {
    const int s = i >> 11, rem = i & 2047;
    const int nt = rem >> 9, d = rem & 511, lane_d = d >> 4, e = d & 15;
    const int kk = s * 32 + b_k_of(lane_d, e);
    const int n = nt * 16 + (lane_d & 15);
    float v = 0.f;
    if (kk < K) v = Wt[n * K + kk];
    BsF[i] = (_Float16)v;
  }
  __syncthreads();
  const int nt = wave;
  v8f acc = {};
#pragma unroll
  for (int s = 0; s < 2; ++s) {
    const v16h a  = *reinterpret_cast<const v16h*>(&AsF[s * 512 + lane * 16]);
    const v16h bf = *reinterpret_cast<const v16h*>(&BsF[(s * 4 + nt) * 512 + lane * 16]);
    acc = WMMA_F16(a, bf, acc);
  }
  const int n = nt * 16 + (lane & 15), half = lane >> 4;
#pragma unroll
  for (int r8 = 0; r8 < 8; ++r8) {
    const int row = row0 + r8 + 8 * half;
    if (row < rows) Y[(size_t)row * 64 + n] = acc[r8] + (bias ? bias[n] : 0.f);
  }
}

// ---------------------------------------------------------------------------
// SAGE combine + gated blend:
//   new = (sums/max(cnt,1)) @ Wl^T + bl + xd @ Wr^T ; Y = (1-a)*xd + a*new
// ---------------------------------------------------------------------------
__global__ __launch_bounds__(128) void sage_blend_kernel(
    const float* __restrict__ sums, const float* __restrict__ cnt,
    const float* __restrict__ xd, const float* __restrict__ Wl,
    const float* __restrict__ bl, const float* __restrict__ Wr,
    const float* __restrict__ theta, float* __restrict__ Y, int rows) {
  __shared__ __align__(32) _Float16 AmF[2 * 512], AxF[2 * 512];
  __shared__ __align__(32) _Float16 BlF[2 * 4 * 512], BrF[2 * 4 * 512];
  __shared__ float Xd32[16 * 64];
  const int tid = threadIdx.x, lane = tid & 31, wave = tid >> 5;
  const int row0 = blockIdx.x * 16;
  for (int i = tid; i < 2 * 512; i += 128) {
    const int s = i >> 9, d = i & 511, lane_d = d >> 4, e = d & 15;
    const int m = lane_d & 15;
    const int kk = s * 32 + a_k_of(lane_d, e);
    float mv = 0.f, xv = 0.f;
    const int row = row0 + m;
    if (row < rows && kk < 64) {
      float c = cnt[row]; if (c < 1.f) c = 1.f;
      mv = sums[(size_t)row * 64 + kk] / c;
      xv = xd[(size_t)row * 64 + kk];
    }
    AmF[i] = (_Float16)mv;
    AxF[i] = (_Float16)xv;
  }
  for (int i = tid; i < 16 * 64; i += 128) {
    const int r = i >> 6, c = i & 63;
    Xd32[i] = (row0 + r < rows) ? xd[(size_t)(row0 + r) * 64 + c] : 0.f;
  }
  for (int i = tid; i < 2 * 4 * 512; i += 128) {
    const int s = i >> 11, rem = i & 2047;
    const int nt = rem >> 9, d = rem & 511, lane_d = d >> 4, e = d & 15;
    const int kk = s * 32 + b_k_of(lane_d, e);
    const int n = nt * 16 + (lane_d & 15);
    BlF[i] = (_Float16)Wl[n * 64 + kk];
    BrF[i] = (_Float16)Wr[n * 64 + kk];
  }
  __syncthreads();
  const int nt = wave;
  v8f acc = {};
#pragma unroll
  for (int s = 0; s < 2; ++s) {
    const v16h a  = *reinterpret_cast<const v16h*>(&AmF[s * 512 + lane * 16]);
    const v16h bf = *reinterpret_cast<const v16h*>(&BlF[(s * 4 + nt) * 512 + lane * 16]);
    acc = WMMA_F16(a, bf, acc);
  }
#pragma unroll
  for (int s = 0; s < 2; ++s) {
    const v16h a  = *reinterpret_cast<const v16h*>(&AxF[s * 512 + lane * 16]);
    const v16h bf = *reinterpret_cast<const v16h*>(&BrF[(s * 4 + nt) * 512 + lane * 16]);
    acc = WMMA_F16(a, bf, acc);
  }
  const float aG = dsigmoid(theta[0]);
  const int n = nt * 16 + (lane & 15), half = lane >> 4;
#pragma unroll
  for (int r8 = 0; r8 < 8; ++r8) {
    const int m = r8 + 8 * half, row = row0 + m;
    if (row < rows) {
      const float newv = acc[r8] + bl[n];
      Y[(size_t)row * 64 + n] = (1.f - aG) * Xd32[m * 64 + n] + aG * newv;
    }
  }
}

// --------------------------- edge / elementwise ----------------------------
__global__ void edge_accum_kernel(const int* __restrict__ src, const int* __restrict__ dst,
                                  const float* __restrict__ xs, float* __restrict__ sums,
                                  float* __restrict__ cnt, int E, int T1, int rowS, int rowD) {
  const long long i = (long long)blockIdx.x * blockDim.x + threadIdx.x;
  if (i >= (long long)E * T1) return;
  const int e = (int)(i % E), t = (int)(i / E);
  const int sr = t * rowS + src[e], dr = t * rowD + dst[e];
  for (int c = 0; c < 64; ++c)
    atomicAdd(&sums[(size_t)dr * 64 + c], xs[(size_t)sr * 64 + c]);
  atomicAdd(&cnt[dr], 1.0f);
}

__global__ void deg_accum_kernel(const int* __restrict__ dst, float* __restrict__ deg,
                                 int E, int T1, int rowD) {
  const long long i = (long long)blockIdx.x * blockDim.x + threadIdx.x;
  if (i >= (long long)E * T1) return;
  const int e = (int)(i % E), t = (int)(i / E);
  atomicAdd(&deg[t * rowD + dst[e]], 1.0f);
}

__global__ void gcn_agg_kernel(const int* __restrict__ src, const int* __restrict__ dst,
                               const float* __restrict__ xw, const float* __restrict__ deg,
                               float* __restrict__ agg, int E, int T1, int rowN) {
  const long long i = (long long)blockIdx.x * blockDim.x + threadIdx.x;
  if (i >= (long long)E * T1) return;
  const int e = (int)(i % E), t = (int)(i / E);
  const int sr = t * rowN + src[e], dr = t * rowN + dst[e];
  const float w = rsqrtf(deg[sr] + 1.0f) * rsqrtf(deg[dr] + 1.0f);
  for (int c = 0; c < 64; ++c)
    atomicAdd(&agg[(size_t)dr * 64 + c], xw[(size_t)sr * 64 + c] * w);
}

__global__ void to_blend_kernel(float* __restrict__ xr, const float* __restrict__ tmsg,
                                const float* __restrict__ omsg, const float* __restrict__ thT,
                                const float* __restrict__ thO, int Nr, long long total) {
  const long long i = (long long)blockIdx.x * blockDim.x + threadIdx.x;
  if (i >= total) return;
  const int c = (int)(i & 63);
  const long long r = i >> 6;
  const int g = (int)(r / Nr);
  const float at = dsigmoid(thT[0]), ao = dsigmoid(thO[0]);
  float v = xr[i];
  v = (1.f - at) * v + at * tmsg[g * 64 + c];
  v = (1.f - ao) * v + ao * omsg[g * 64 + c];
  xr[i] = v;
}

__global__ void gcn_combine_kernel(const float* __restrict__ aggh, const float* __restrict__ xwh,
                                   const float* __restrict__ degh, const float* __restrict__ bh,
                                   const float* __restrict__ aggv, const float* __restrict__ xwv,
                                   const float* __restrict__ degv, const float* __restrict__ bv,
                                   float* __restrict__ out, long long total) {
  const long long i = (long long)blockIdx.x * blockDim.x + threadIdx.x;
  if (i >= total) return;
  const int c = (int)(i & 63);
  const long long r = i >> 6;
  const float dh = degh[r] + 1.0f, dv = degv[r] + 1.0f;
  float v = aggh[i] + xwh[i] / dh + bh[c] + aggv[i] + xwv[i] / dv + bv[c];
  out[i] = v > 0.f ? v : 0.f;
}

// ---------------------------------------------------------------------------
extern "C" void kernel_launch(void* const* d_in, const int* in_sizes, int n_in,
                              void* d_out, int out_size, void* d_ws, size_t ws_size,
                              hipStream_t stream) {
  (void)n_in; (void)out_size; (void)ws_size;
  auto F = [&](int i) { return (const float*)d_in[i]; };
  auto I = [&](int i) { return (const int*)d_in[i]; };
  auto cdiv = [](long long a, long long b) { return (unsigned)((a + b - 1) / b); };

  const int B = 4, T = 48, T1 = 46, T2 = 44;
  const int Nr = 256, Nd = 1024, Np = 4096;
  const int P1r = T1 * B * Nr, P1d = T1 * B * Nd, P1p = T1 * B * Np, P1t = T1 * B;
  const int P2r = T2 * B * Nr, P2d = T2 * B * Nd, P2p = T2 * B * Np, P2t = T2 * B;

  float* w = (float*)d_ws;
  size_t off = 0;
  auto alloc = [&](size_t nf) { float* p = w + off; off += nf; return p; };
  float* xm_room = alloc((size_t)P1r * 64);
  float* xm_dev  = alloc((size_t)P1d * 64);
  float* xm_prop = alloc((size_t)P1p * 64);
  float* xm_time = alloc((size_t)P1t * 32);
  float* xm_out  = alloc((size_t)P1t * 16);
  float* xdev2   = alloc((size_t)P1d * 64);
  float* xroom   = alloc((size_t)P1r * 64);
  float* tmsg    = alloc((size_t)P1t * 64);
  float* omsg    = alloc((size_t)P1t * 64);
  float* sums_d  = alloc((size_t)P1d * 64);
  float* cnt_d   = alloc((size_t)P1d);
  float* sums_r  = alloc((size_t)P1r * 64);
  float* cnt_r   = alloc((size_t)P1r);
  float* degh    = alloc((size_t)P1r);
  float* degv    = alloc((size_t)P1r);
  float* xwh     = alloc((size_t)P1r * 64);
  float* xwv     = alloc((size_t)P1r * 64);
  float* aggh    = alloc((size_t)P1r * 64);
  float* aggv    = alloc((size_t)P1r * 64);

  hipMemsetAsync(sums_d, 0, (size_t)P1d * 64 * sizeof(float), stream);
  hipMemsetAsync(cnt_d, 0, (size_t)P1d * sizeof(float), stream);
  hipMemsetAsync(sums_r, 0, (size_t)P1r * 64 * sizeof(float), stream);
  hipMemsetAsync(cnt_r, 0, (size_t)P1r * sizeof(float), stream);
  hipMemsetAsync(degh, 0, (size_t)P1r * sizeof(float), stream);
  hipMemsetAsync(degv, 0, (size_t)P1r * sizeof(float), stream);
  hipMemsetAsync(aggh, 0, (size_t)P1r * 64 * sizeof(float), stream);
  hipMemsetAsync(aggv, 0, (size_t)P1r * 64 * sizeof(float), stream);

  // ---- tconv1 per type (params start at 9, 6 per type) ----
  tconv1_kernel<32, 64, 96><<<cdiv(P1r, 16), 128, 0, stream>>>(F(0), F(9),  F(10), xm_room, B, T, Nr, P1r);
  tconv1_kernel<32, 64, 96><<<cdiv(P1d, 16), 128, 0, stream>>>(F(1), F(15), F(16), xm_dev,  B, T, Nd, P1d);
  tconv1_kernel<32, 64, 96><<<cdiv(P1p, 16), 128, 0, stream>>>(F(2), F(21), F(22), xm_prop, B, T, Np, P1p);
  tconv1_kernel<16, 32, 64><<<cdiv(P1t, 16), 128, 0, stream>>>(F(3), F(27), F(28), xm_time, B, T, 1, P1t);
  tconv1_kernel<8, 16, 32><<<cdiv(P1t, 16), 128, 0, stream>>>(F(4), F(33), F(34), xm_out, B, T, 1, P1t);

  // ---- SAGE 1: property -> device ----
  const int Ep = in_sizes[5] / 2, Ed = in_sizes[6] / 2, Er = in_sizes[7] / 2;
  edge_accum_kernel<<<cdiv((long long)Ep * T1, 256), 256, 0, stream>>>(
      I(5), I(5) + Ep, xm_prop, sums_d, cnt_d, Ep, T1, B * Np, B * Nd);
  sage_blend_kernel<<<cdiv(P1d, 16), 128, 0, stream>>>(
      sums_d, cnt_d, xm_dev, F(39), F(40), F(41), F(53), xdev2, P1d);

  // ---- SAGE 2: device -> room ----
  edge_accum_kernel<<<cdiv((long long)Ed * T1, 256), 256, 0, stream>>>(
      I(6), I(6) + Ed, xdev2, sums_r, cnt_r, Ed, T1, B * Nd, B * Nr);
  sage_blend_kernel<<<cdiv(P1r, 16), 128, 0, stream>>>(
      sums_r, cnt_r, xm_room, F(42), F(43), F(44), F(54), xroom, P1r);

  // ---- time / outside projections + blends ----
  gemm_nt_bias_kernel<<<cdiv(P1t, 16), 128, 0, stream>>>(xm_time, F(45), F(46), tmsg, P1t, 32);
  gemm_nt_bias_kernel<<<cdiv(P1t, 16), 128, 0, stream>>>(xm_out,  F(47), F(48), omsg, P1t, 16);
  to_blend_kernel<<<cdiv((long long)P1r * 64, 256), 256, 0, stream>>>(
      xroom, tmsg, omsg, F(55), F(56), Nr, (long long)P1r * 64);

  // ---- GCN h + v over rooms ----
  gemm_nt_bias_kernel<<<cdiv(P1r, 16), 128, 0, stream>>>(xroom, F(49), nullptr, xwh, P1r, 64);
  gemm_nt_bias_kernel<<<cdiv(P1r, 16), 128, 0, stream>>>(xroom, F(51), nullptr, xwv, P1r, 64);
  deg_accum_kernel<<<cdiv((long long)Er * T1, 256), 256, 0, stream>>>(I(7) + Er, degh, Er, T1, B * Nr);
  deg_accum_kernel<<<cdiv((long long)Er * T1, 256), 256, 0, stream>>>(I(8) + Er, degv, Er, T1, B * Nr);
  gcn_agg_kernel<<<cdiv((long long)Er * T1, 256), 256, 0, stream>>>(
      I(7), I(7) + Er, xwh, degh, aggh, Er, T1, B * Nr);
  gcn_agg_kernel<<<cdiv((long long)Er * T1, 256), 256, 0, stream>>>(
      I(8), I(8) + Er, xwv, degv, aggv, Er, T1, B * Nr);
  gcn_combine_kernel<<<cdiv((long long)P1r * 64, 256), 256, 0, stream>>>(
      aggh, xwh, degh, F(50), aggv, xwv, degv, F(52), xroom, (long long)P1r * 64);

  // ---- tconv2 + LayerNorm into d_out (room, device, property, time, outside) ----
  float* out = (float*)d_out;
  float* o_room = out;
  float* o_dev  = out + (size_t)B * 64 * T2 * Nr;
  float* o_prop = o_dev + (size_t)B * 64 * T2 * Nd;
  float* o_time = o_prop + (size_t)B * 64 * T2 * Np;
  float* o_outs = o_time + (size_t)B * 64 * T2;
  tconv2_ln_kernel<64, 192><<<cdiv(P2r, 16), 128, 0, stream>>>(xroom,   F(11), F(12), F(13), F(14), o_room, B, Nr, T2, P2r);
  tconv2_ln_kernel<64, 192><<<cdiv(P2d, 16), 128, 0, stream>>>(xdev2,   F(17), F(18), F(19), F(20), o_dev,  B, Nd, T2, P2d);
  tconv2_ln_kernel<64, 192><<<cdiv(P2p, 16), 128, 0, stream>>>(xm_prop, F(23), F(24), F(25), F(26), o_prop, B, Np, T2, P2p);
  tconv2_ln_kernel<32, 96><<<cdiv(P2t, 16), 128, 0, stream>>>(xm_time, F(29), F(30), F(31), F(32), o_time, B, 1, T2, P2t);
  tconv2_ln_kernel<16, 64><<<cdiv(P2t, 16), 128, 0, stream>>>(xm_out,  F(35), F(36), F(37), F(38), o_outs, B, 1, T2, P2t);
}
